// SplitPointNet_52441550684642
// MI455X (gfx1250) — compile-verified, matrix operand order checked by reading, not run on hardware
//
#include <hip/hip_runtime.h>
#include <hip/hip_bf16.h>

typedef _Float16 v16h __attribute__((ext_vector_type(16)));
typedef _Float16 v8h  __attribute__((ext_vector_type(8)));
typedef float    v8f  __attribute__((ext_vector_type(8)));

// ---------------- layer / buffer geometry (shared host+device) ----------------
// 9 layers per branch: n1[0],n1[1],n2[0],n2[1],n3[0],n3[1],n3[2],fc1,fc2
constexpr int L_COUTP[9] = {32, 64, 64, 128, 128, 256, 512, 256, 32};   // padded rows
constexpr int L_CIN  [9] = {6, 32, 67, 64, 131, 128, 256, 512, 256};
constexpr int L_KPAD [9] = {32, 32, 96, 64, 160, 128, 256, 512, 256};
constexpr int L_WOFF [9] = {0, 1024, 3072, 9216, 17408, 37888, 70656, 201728, 332800};
constexpr int WBH        = 340992;              // halfs per branch of fp16 weights

constexpr size_t W_OFF_B   = 0;
constexpr size_t W_BYTES   = 6ull * WBH * 2;                 // 4,091,904
constexpr size_t SB_OFF_B  = W_OFF_B + W_BYTES;              // scale/bias: 6*9*1024 f32
constexpr size_t SB_BYTES  = 6ull * 9 * 1024 * 4;            // 221,184
constexpr size_t IDX_OFF_B = SB_OFF_B + SB_BYTES;            // split idx: 6*512 i32
constexpr size_t IDX_BYTES = 6ull * 512 * 4;
constexpr size_t STG_OFF_B = IDX_OFF_B + IDX_BYTES;          // stage: 6*128*18 f32

// dynamic LDS layout (bytes) for branch kernel (worst case branch 5)
constexpr int PTS_OFF  = 0;          // [512][8] f32           = 16384
constexpr int NBR1_OFF = 16384;      // [128][64] u16          = 16384
constexpr int XF_OFF   = 32768;      // [128][96] f16          = 24576
constexpr int NBR2_OFF = 57344;      // [64][64] u16           =  8192
constexpr int X2_OFF   = 65536;      // [64][128] f16          = 16384
constexpr int Y_OFF    = 81920;      // [512] f32              =  2048
constexpr int H_OFF    = 83968;      // [256] f32              =  1024
constexpr int TA_OFF   = 84992;      // 32 KB tile buffer A
constexpr int TB_OFF   = 117760;     // 32 KB tile buffer B
constexpr int SMEM_BYTES = 150528;   // 147 KB -> 2 workgroups per 320 KB WGP

constexpr float SPLIT_OFFS[6][3] = {
  {-0.1f, 0.5f, 0.0f}, {0.3f, 0.3f, 0.0f}, {0.35f, 0.0f, 0.0f},
  {0.3f, -0.3f, 0.0f}, {0.25f, -0.5f, 0.0f}, {-0.25f, -0.25f, -0.25f}};

struct PtrPack50 { const float* p[50]; };

// ---------------- WMMA fragment helpers ----------------
__device__ inline v16h load_frag_A(const _Float16* base, int strideH, int mt, int kt, int lane) {
  int r = mt * 16 + (lane & 15);
  const _Float16* p = base + r * strideH + kt * 32 + ((lane >> 4) << 3);
  const v8h* q = (const v8h*)p;
  v8h lo = q[0], hi = q[2];            // +16 halfs
  v16h a;
#pragma unroll
  for (int j = 0; j < 8; ++j) { a[j] = lo[j]; a[j + 8] = hi[j]; }
  return a;
}
__device__ inline v16h load_frag_B(const _Float16* w, int kpad, int nt, int kt, int lane) {
  int n = nt * 16 + (lane & 15);
  const _Float16* p = w + (size_t)n * kpad + kt * 32 + ((lane >> 4) << 3);
  const v8h* q = (const v8h*)p;
  v8h lo = q[0], hi = q[2];
  v16h b;
#pragma unroll
  for (int j = 0; j < 8; ++j) { b[j] = lo[j]; b[j + 8] = hi[j]; }
  return b;
}

// GEMM tile: out[ROWS][COUT] = relu(scale * (A[ROWS][KPAD] @ W[COUT][KPAD]^T) + bias), fp16 out
template <int ROWS, int KPAD, int COUT>
__device__ inline void gemm_bn_relu(const _Float16* __restrict__ A, const _Float16* __restrict__ W,
                                    const float* __restrict__ scale, const float* __restrict__ bias,
                                    _Float16* __restrict__ out, int outStride, int lane, int wave) {
  constexpr int MT = ROWS / 16, NT = COUT / 16, KT = KPAD / 32;
  for (int tile = wave; tile < MT * NT; tile += 8) {
    int mt = tile % MT, nt = tile / MT;
    v8f acc = {};
#pragma unroll
    for (int kt = 0; kt < KT; ++kt) {
      v16h a = load_frag_A(A, KPAD, mt, kt, lane);
      v16h b = load_frag_B(W, KPAD, nt, kt, lane);
      acc = __builtin_amdgcn_wmma_f32_16x16x32_f16(false, a, false, b, (short)0, acc, false, false);
    }
    int col = nt * 16 + (lane & 15);
    float s = scale[col], bb = bias[col];
    int rbase = mt * 16 + ((lane >> 4) << 3);
#pragma unroll
    for (int v = 0; v < 8; ++v) {
      float y = fmaxf(acc[v] * s + bb, 0.0f);
      out[(rbase + v) * outStride + col] = (_Float16)y;
    }
  }
}

// Same GEMM, but epilogue fuses max-pool over all ROWS into ypool[COUT] (LDS, f32 >= 0).
// Valid because ReLU output >= 0 => float bits are monotonic as ints (ds_max on ints).
template <int ROWS, int KPAD, int COUT>
__device__ inline void gemm_bn_relu_maxpool(const _Float16* __restrict__ A,
                                            const _Float16* __restrict__ W,
                                            const float* __restrict__ scale,
                                            const float* __restrict__ bias,
                                            float* __restrict__ ypool, int lane, int wave) {
  constexpr int MT = ROWS / 16, NT = COUT / 16, KT = KPAD / 32;
  for (int tile = wave; tile < MT * NT; tile += 8) {
    int mt = tile % MT, nt = tile / MT;
    v8f acc = {};
#pragma unroll
    for (int kt = 0; kt < KT; ++kt) {
      v16h a = load_frag_A(A, KPAD, mt, kt, lane);
      v16h b = load_frag_B(W, KPAD, nt, kt, lane);
      acc = __builtin_amdgcn_wmma_f32_16x16x32_f16(false, a, false, b, (short)0, acc, false, false);
    }
    int col = nt * 16 + (lane & 15);
    float s = scale[col], bb = bias[col];
    float m = 0.0f;
#pragma unroll
    for (int v = 0; v < 8; ++v) m = fmaxf(m, fmaxf(acc[v] * s + bb, 0.0f));
    atomicMax((int*)(ypool + col), __float_as_int(m));
  }
}

// ---------------- wave-parallel KNN-64 (order-insensitive downstream: max-pool) ----------------
template <int NSLOTS>  // np = NSLOTS*32 points, centers are points [0, ncenters)
__device__ inline void knn_select(const float* __restrict__ pts8, int ncenters, float ballR,
                                  unsigned short* __restrict__ nbr, int tid) {
  int lane = tid & 31, wave = tid >> 5;
  for (int c = wave; c < ncenters; c += 8) {
    float cx = pts8[c * 8 + 0], cy = pts8[c * 8 + 1], cz = pts8[c * 8 + 2];
    float d2l[NSLOTS];
#pragma unroll
    for (int s = 0; s < NSLOTS; ++s) {
      int p = s * 32 + lane;
      float dx = pts8[p * 8 + 0] - cx, dy = pts8[p * 8 + 1] - cy, dz = pts8[p * 8 + 2] - cz;
      d2l[s] = dx * dx + dy * dy + dz * dz;
    }
    for (int k = 0; k < 64; ++k) {
      unsigned best = 0xFFFFFFFFu;
#pragma unroll
      for (int s = 0; s < NSLOTS; ++s) {
        unsigned key = (__float_as_uint(d2l[s]) & 0xFFFFFE00u) | (unsigned)(s * 32 + lane);
        best = best < key ? best : key;
      }
#pragma unroll
      for (int off = 16; off > 0; off >>= 1) {
        unsigned o = (unsigned)__shfl_xor((int)best, off, 32);
        best = best < o ? best : o;
      }
      unsigned pidx = best & 511u;
      float d2w = __uint_as_float(best & 0xFFFFFE00u);
      unsigned short sel = (d2w > ballR) ? (unsigned short)c : (unsigned short)pidx;
      if (lane == 0) nbr[c * 64 + k] = sel;
      int slot = (int)(pidx >> 5);
      if ((pidx & 31u) == (unsigned)lane) {
#pragma unroll
        for (int s = 0; s < NSLOTS; ++s)
          if (s == slot) d2l[s] = 3.0e38f;
      }
    }
  }
}

// ---------------- kernel 1: weight fp16 pack + BN fold ----------------
__global__ __launch_bounds__(256) void prep_kernel(PtrPack50 pk, _Float16* wh, float* sb,
                                                   int branch, int nout) {
  int l = blockIdx.x, tid = threadIdx.x;
  int rows = L_COUTP[l], cin = L_CIN[l], kpad = L_KPAD[l];
  int cout = (l == 8) ? nout : rows;
  const float* W = pk.p[(l == 8) ? 48 : l * 6 + 0];
  const float* b = pk.p[(l == 8) ? 49 : l * 6 + 1];
  _Float16* wdst = wh + (size_t)branch * WBH + L_WOFF[l];
  int total = rows * kpad;
  for (int i = tid; i < total; i += 256) {
    int o = i / kpad, k = i - o * kpad;
    float w = (o < cout && k < cin) ? W[o * cin + k] : 0.0f;
    wdst[i] = (_Float16)w;
  }
  float* sbp = sb + ((size_t)branch * 9 + l) * 1024;
  for (int o = tid; o < 512; o += 256) {
    float sc = 1.0f, bi = 0.0f;
    if (o < cout) {
      if (l == 8) { bi = b[o]; }
      else {
        const float* g = pk.p[l * 6 + 2];
        const float* bt = pk.p[l * 6 + 3];
        const float* m = pk.p[l * 6 + 4];
        const float* v = pk.p[l * 6 + 5];
        float s = g[o] * rsqrtf(v[o] + 1e-5f);
        sc = s; bi = b[o] * s + bt[o] - m[o] * s;
      }
    }
    sbp[o] = sc; sbp[512 + o] = bi;
  }
}

// ---------------- kernel 2: batch-0 split (center + bitonic top-k indices) ----------------
__global__ __launch_bounds__(256) void split_kernel(const float* __restrict__ points,
                                                    int* __restrict__ idxout) {
  __shared__ float red[768];
  __shared__ unsigned keys[2048];
  __shared__ float ctr[3];
  int br = blockIdx.x, t = threadIdx.x;
  float sx = 0, sy = 0, sz = 0;
  for (int n = t; n < 2048; n += 256) {
    const float* p = points + n * 6;
    sx += p[0]; sy += p[1]; sz += p[2];
  }
  red[t] = sx; red[256 + t] = sy; red[512 + t] = sz;
  __syncthreads();
  for (int s = 128; s > 0; s >>= 1) {
    if (t < s) { red[t] += red[t + s]; red[256 + t] += red[256 + t + s]; red[512 + t] += red[512 + t + s]; }
    __syncthreads();
  }
  if (t == 0) {
    ctr[0] = red[0] / 2048.0f - SPLIT_OFFS[br][0];
    ctr[1] = red[256] / 2048.0f - SPLIT_OFFS[br][1];
    ctr[2] = red[512] / 2048.0f - SPLIT_OFFS[br][2];
  }
  __syncthreads();
  for (int n = t; n < 2048; n += 256) {
    const float* p = points + n * 6;
    float dx = p[0] - ctr[0], dy = p[1] - ctr[1], dz = p[2] - ctr[2];
    float d2 = dx * dx + dy * dy + dz * dz;
    keys[n] = (__float_as_uint(d2) & 0xFFFFF800u) | (unsigned)n;
  }
  __syncthreads();
  for (int k = 2; k <= 2048; k <<= 1) {
    for (int j = k >> 1; j > 0; j >>= 1) {
      for (int i = t; i < 2048; i += 256) {
        int ixj = i ^ j;
        if (ixj > i) {
          bool up = ((i & k) == 0);
          unsigned a = keys[i], bb = keys[ixj];
          if ((a > bb) == up) { keys[i] = bb; keys[ixj] = a; }
        }
      }
      __syncthreads();
    }
  }
  int cnt = (br == 5) ? 512 : 256;
  for (int n = t; n < cnt; n += 256) idxout[br * 512 + n] = (int)(keys[n] & 2047u);
}

// ---------------- kernel 3: fused per-(branch,batch) PointNet branch ----------------
__global__ __launch_bounds__(256) void branch_kernel(const float* __restrict__ points,
                                                     const _Float16* __restrict__ wh,
                                                     const float* __restrict__ sb,
                                                     const int* __restrict__ idxsplit,
                                                     float* __restrict__ stage) {
  extern __shared__ char smem[];
  int br = blockIdx.y, batch = blockIdx.x, tid = threadIdx.x;
  int lane = tid & 31, wave = tid >> 5;
  int NP = (br == 5) ? 512 : 256;
  int SN1 = (br == 5) ? 128 : 64;
  int SN2 = (br == 5) ? 64 : 32;
  int NOUT = (br == 5) ? 18 : 12;

  float* pts = (float*)(smem + PTS_OFF);
  unsigned short* nbr1 = (unsigned short*)(smem + NBR1_OFF);
  _Float16* xfeat = (_Float16*)(smem + XF_OFF);
  unsigned short* nbr2 = (unsigned short*)(smem + NBR2_OFF);
  _Float16* x2feat = (_Float16*)(smem + X2_OFF);
  float* ybuf = (float*)(smem + Y_OFF);
  float* hbuf = (float*)(smem + H_OFF);
  _Float16* TA = (_Float16*)(smem + TA_OFF);
  _Float16* TB = (_Float16*)(smem + TB_OFF);

  const _Float16* wB = wh + (size_t)br * WBH;
  const float* sbB = sb + (size_t)br * 9 * 1024;

  // warm the cache hierarchy with this branch's weights (global_prefetch_b8)
  for (int off = tid * 64; off < WBH; off += 256 * 64)
    __builtin_prefetch(wB + off, 0, 1);

  // zero xfeat pad region + pooled-y accumulator
  for (int i = tid; i < 128 * 96; i += 256) xfeat[i] = (_Float16)0.0f;
  for (int i = tid; i < 512; i += 256) ybuf[i] = 0.0f;
  // gather part points
  for (int n = tid; n < NP; n += 256) {
    int g = idxsplit[br * 512 + n];
    const float* src = points + ((size_t)batch * 2048 + g) * 6;
    float* row = pts + n * 8;
#pragma unroll
    for (int j = 0; j < 6; ++j) row[j] = src[j];
    row[6] = 0.0f; row[7] = 0.0f;
  }
  __syncthreads();

  // ---- grouping 1 ----
  if (NP == 512) knn_select<16>(pts, SN1, 0.015f, nbr1, tid);
  else           knn_select<8>(pts, SN1, 0.015f, nbr1, tid);
  __syncthreads();

  // ---- MLP1: (6->32->64) per 4-center tile of 256 positions, max-pool over k ----
  for (int g = 0; g < SN1 / 4; ++g) {
    {
      int cl = tid >> 6, k = tid & 63;
      int c = g * 4 + cl;
      int nb = nbr1[c * 64 + k];
      const float* pr = pts + nb * 8;
      float cx = pts[c * 8], cy = pts[c * 8 + 1], cz = pts[c * 8 + 2];
      _Float16* row = TA + tid * 32;
      row[0] = (_Float16)(pr[0] - cx); row[1] = (_Float16)(pr[1] - cy); row[2] = (_Float16)(pr[2] - cz);
      row[3] = (_Float16)pr[3]; row[4] = (_Float16)pr[4]; row[5] = (_Float16)pr[5];
#pragma unroll
      for (int j = 6; j < 32; ++j) row[j] = (_Float16)0.0f;
    }
    __syncthreads();
    gemm_bn_relu<256, 32, 32>(TA, wB + L_WOFF[0], sbB + 0 * 1024, sbB + 0 * 1024 + 512, TB, 32, lane, wave);
    __syncthreads();
    gemm_bn_relu<256, 32, 64>(TB, wB + L_WOFF[1], sbB + 1 * 1024, sbB + 1 * 1024 + 512, TA, 64, lane, wave);
    __syncthreads();
    {
      int cl = tid >> 6, ch = tid & 63;
      float m = -3.0e38f;
#pragma unroll 8
      for (int k = 0; k < 64; ++k) m = fmaxf(m, (float)TA[(cl * 64 + k) * 64 + ch]);
      xfeat[(g * 4 + cl) * 96 + 3 + ch] = (_Float16)m;
    }
    __syncthreads();
  }

  // ---- grouping 2 (over the SN1 centers, geometry from fp32 pts) ----
  if (SN1 == 128) knn_select<4>(pts, SN2, 0.04f, nbr2, tid);
  else            knn_select<2>(pts, SN2, 0.04f, nbr2, tid);
  __syncthreads();

  // ---- MLP2: (67->64->128) per 2-center tile of 128 positions, max-pool ----
  for (int g = 0; g < SN2 / 2; ++g) {
    {
      int r = tid >> 1, half = tid & 1;        // 2 threads per row, 48 halfs each
      int c = g * 2 + (r >> 6), k = r & 63;
      int nb = nbr2[c * 64 + k];
      _Float16* row = TA + r * 96 + half * 48;
      const _Float16* xf = xfeat + nb * 96;
#pragma unroll
      for (int j = 0; j < 48; ++j) {
        int jj = half * 48 + j;
        _Float16 v;
        if (jj < 3)       v = (_Float16)(pts[nb * 8 + jj] - pts[c * 8 + jj]);
        else if (jj < 67) v = xf[jj];
        else              v = (_Float16)0.0f;
        row[j] = v;
      }
    }
    __syncthreads();
    gemm_bn_relu<128, 96, 64>(TA, wB + L_WOFF[2], sbB + 2 * 1024, sbB + 2 * 1024 + 512, TB, 64, lane, wave);
    __syncthreads();
    gemm_bn_relu<128, 64, 128>(TB, wB + L_WOFF[3], sbB + 3 * 1024, sbB + 3 * 1024 + 512, TA, 128, lane, wave);
    __syncthreads();
    {
      int cl = tid >> 7, ch = tid & 127;
      float m = -3.0e38f;
#pragma unroll 8
      for (int k = 0; k < 64; ++k) m = fmaxf(m, (float)TA[(cl * 64 + k) * 128 + ch]);
      x2feat[(g * 2 + cl) * 128 + ch] = (_Float16)m;
    }
    __syncthreads();
  }

  // ---- MLP3: (131->128->256->512) over SN2 rows, final layer pools into ybuf ----
  for (int i = tid; i < SN2 * 160; i += 256) {
    int r = i / 160, j = i - r * 160;
    _Float16 v = (_Float16)0.0f;
    if (j < 3) v = (_Float16)pts[r * 8 + j];
    else if (j < 131) v = x2feat[r * 128 + (j - 3)];
    TA[i] = v;
  }
  __syncthreads();
  if (SN2 == 64) {
    gemm_bn_relu<64, 160, 128>(TA, wB + L_WOFF[4], sbB + 4 * 1024, sbB + 4 * 1024 + 512, TB, 128, lane, wave);
    __syncthreads();
    gemm_bn_relu<64, 128, 256>(TB, wB + L_WOFF[5], sbB + 5 * 1024, sbB + 5 * 1024 + 512, TA, 256, lane, wave);
    __syncthreads();
    gemm_bn_relu_maxpool<64, 256, 512>(TA, wB + L_WOFF[6], sbB + 6 * 1024, sbB + 6 * 1024 + 512, ybuf, lane, wave);
  } else {
    gemm_bn_relu<32, 160, 128>(TA, wB + L_WOFF[4], sbB + 4 * 1024, sbB + 4 * 1024 + 512, TB, 128, lane, wave);
    __syncthreads();
    gemm_bn_relu<32, 128, 256>(TB, wB + L_WOFF[5], sbB + 5 * 1024, sbB + 5 * 1024 + 512, TA, 256, lane, wave);
    __syncthreads();
    gemm_bn_relu_maxpool<32, 256, 512>(TA, wB + L_WOFF[6], sbB + 6 * 1024, sbB + 6 * 1024 + 512, ybuf, lane, wave);
  }
  __syncthreads();

  // ---- fc1 (512->256, bn, relu) ----
  {
    const _Float16* w = wB + L_WOFF[7] + (size_t)tid * 512;
    float s = 0.0f;
#pragma unroll 8
    for (int k = 0; k < 512; ++k) s += ybuf[k] * (float)w[k];
    const float* sc = sbB + 7 * 1024;
    hbuf[tid] = fmaxf(s * sc[tid] + sc[512 + tid], 0.0f);
  }
  __syncthreads();
  // ---- fc2 (256->NOUT) ----
  if (tid < NOUT) {
    const _Float16* w = wB + L_WOFF[8] + (size_t)tid * 256;
    float s = 0.0f;
#pragma unroll 8
    for (int k = 0; k < 256; ++k) s += hbuf[k] * (float)w[k];
    s += (sbB + 8 * 1024)[512 + tid];
    stage[((size_t)br * 128 + batch) * 18 + tid] = s;
  }
}

// ---------------- kernel 4: combine 6 branch outputs into (B,63) ----------------
__global__ __launch_bounds__(64) void combine_kernel(const float* __restrict__ stage,
                                                     float* __restrict__ out) {
  int batch = blockIdx.x, t = threadIdx.x;
  if (t >= 63) return;
  auto X = [&](int b, int j) { return stage[((size_t)b * 128 + batch) * 18 + j]; };
  float r;
  if (t < 3) r = X(5, t);
  else if (t < 18) {
    int i = t / 3 - 1;
    r = 0.8f * X(5, t) + 0.2f * X(i, t - 3 * (i + 1));
  } else {
    int i = (t - 18) / 9, j = (t - 18) % 9;
    r = X(i, 3 + j);
  }
  out[(size_t)batch * 63 + t] = r;
}

// ---------------- host ----------------
extern "C" void kernel_launch(void* const* d_in, const int* in_sizes, int n_in,
                              void* d_out, int out_size, void* d_ws, size_t ws_size,
                              hipStream_t stream) {
  const float* points = (const float*)d_in[0];
  char* ws = (char*)d_ws;
  _Float16* wh = (_Float16*)(ws + W_OFF_B);
  float* sb = (float*)(ws + SB_OFF_B);
  int* idx = (int*)(ws + IDX_OFF_B);
  float* stage = (float*)(ws + STG_OFF_B);

  for (int br = 0; br < 6; ++br) {
    PtrPack50 pk;
    for (int j = 0; j < 50; ++j) pk.p[j] = (const float*)d_in[1 + br * 50 + j];
    prep_kernel<<<9, 256, 0, stream>>>(pk, wh, sb, br, (br == 5) ? 18 : 12);
  }
  split_kernel<<<6, 256, 0, stream>>>(points, idx);

  hipFuncSetAttribute(reinterpret_cast<const void*>(branch_kernel),
                      hipFuncAttributeMaxDynamicSharedMemorySize, SMEM_BYTES);
  branch_kernel<<<dim3(128, 6), 256, SMEM_BYTES, stream>>>(points, wh, sb, idx, stage);

  combine_kernel<<<128, 64, 0, stream>>>(stage, (float*)d_out);
}
